// Encoder_60748017435353
// MI455X (gfx1250) — matrix operand order repaired and checked by reading, not compiled, not run on hardware
//
#include <hip/hip_runtime.h>
#include <math.h>

typedef __attribute__((ext_vector_type(16))) _Float16 v16h;
typedef __attribute__((ext_vector_type(8)))  _Float16 v8h;
typedef __attribute__((ext_vector_type(4)))  _Float16 v4h;
typedef __attribute__((ext_vector_type(2)))  _Float16 v2h;
typedef __attribute__((ext_vector_type(8)))  float    v8f;

#define B_SZ 8
#define T_SZ 24

// ---------------------------------------------------------------- utilities
__global__ void zero_kernel(float* __restrict__ p, long n) {
    long i = (long)blockIdx.x * blockDim.x + threadIdx.x;
    if (i < n) p[i] = 0.f;
}

__global__ void copy_kernel(float* __restrict__ dst,
                            const float* __restrict__ src, long n) {
    long i = (long)blockIdx.x * blockDim.x + threadIdx.x;
    if (i < n) dst[i] = src[i];
}

// -------------------------------------------------- fused BN + 2x2 max pool
__global__ void bnpool_kernel(const float* __restrict__ in, float* __restrict__ out,
                              const float* __restrict__ g, const float* __restrict__ bb,
                              const float* __restrict__ mm, const float* __restrict__ vv,
                              int H, int W, int C, int Ho, int Wo, long total) {
    long idx = (long)blockIdx.x * blockDim.x + threadIdx.x;
    if (idx >= total) return;
    int c = (int)(idx % C); long r = idx / C;
    int xo = (int)(r % Wo); r /= Wo;
    int yo = (int)(r % Ho); r /= Ho;       // r now = b*T + t
    float scale = g[c] / sqrtf(vv[c] + 1e-3f);
    float mean = mm[c], beta = bb[c];
    float best = -INFINITY;
    for (int dy = 0; dy < 2; dy++)
        for (int dx = 0; dx < 2; dx++) {
            int y = yo * 2 + dy, x = xo * 2 + dx;
            if (y < H && x < W) {
                float val = in[((r * H + y) * (long)W + x) * C + c];
                val = (val - mean) * scale + beta;
                best = fmaxf(best, val);
            }
        }
    out[idx] = best;
}

// --------------------------------------------- weight prepack (once/launch)
// Packs Wx (9*Cin x G) and Wh (9*F x G) f32 HWIO weights into WMMA B-fragment
// order with zero padding to 32-row chunks:
//   pw[((kc*NT + nt)*32 + lane)*16 + j]  holds k = kc*32 + j + 16*(lane>>4)
template <int F, int Cin>
__global__ void prepack_weights(const float* __restrict__ wx,
                                const float* __restrict__ wh,
                                _Float16* __restrict__ pw) {
    constexpr int G   = 4 * F;
    constexpr int NT  = G / 16;
    constexpr int K1  = 9 * Cin;
    constexpr int K1P = (K1 + 31) & ~31;
    constexpr int K2  = 9 * F;
    constexpr int K2P = (K2 + 31) & ~31;
    constexpr int KP  = K1P + K2P;
    int idx = blockIdx.x * blockDim.x + threadIdx.x;     // one per (kc,nt,lane)
    if (idx >= (KP / 32) * NT * 32) return;
    int lane = idx & 31; int rest = idx >> 5;
    int nt = rest % NT;  int kc = rest / NT;
    int khalf = lane >> 4, ncol = lane & 15;
    _Float16* dst = pw + (size_t)idx * 16;
#pragma unroll
    for (int j = 0; j < 16; j++) {
        int k = kc * 32 + j + 16 * khalf;
        float v = 0.f;
        if (k < K1)                        v = wx[(long)k * G + nt * 16 + ncol];
        else if (k >= K1P && k - K1P < K2) v = wh[(long)(k - K1P) * G + nt * 16 + ncol];
        dst[j] = (_Float16)v;
    }
}

// ----------------------------------- vectorized f32 -> f16 row staging copy
template <int C, int V>
__device__ __forceinline__ void stage_row(_Float16* __restrict__ dst,
                                          const float* __restrict__ src, bool inb) {
#pragma unroll
    for (int ci = 0; ci < C; ci += V) {
        if constexpr (V == 4) {
            v4h o;
            if (inb) {
                float4 v = *reinterpret_cast<const float4*>(src + ci);
                o[0] = (_Float16)v.x; o[1] = (_Float16)v.y;
                o[2] = (_Float16)v.z; o[3] = (_Float16)v.w;
            } else {
                o[0] = o[1] = o[2] = o[3] = (_Float16)0.f;
            }
            *reinterpret_cast<v4h*>(dst + ci) = o;
        } else if constexpr (V == 2) {
            v2h o;
            if (inb) {
                float2 v = *reinterpret_cast<const float2*>(src + ci);
                o[0] = (_Float16)v.x; o[1] = (_Float16)v.y;
            } else {
                o[0] = o[1] = (_Float16)0.f;
            }
            *reinterpret_cast<v2h*>(dst + ci) = o;
        } else {
            dst[ci] = inb ? (_Float16)src[ci] : (_Float16)0.f;
        }
    }
}

// ------------------------------------------------------- fused ConvLSTM step
// Block = 16 spatial positions x full gate vector G=4F. All M tiles are exact
// multiples of 16 for every layer, so no bounds guards: EXEC stays all-ones.
// Phase 1: cooperative vectorized im2col of x-patch + h-patch into LDS (f16).
// Phase 2: per-wave GEMM over KP chunks: A from LDS (2x ds_load_b128),
//          B from prepacked global (2x global_load_b128), 1 WMMA per chunk.
// Phase 3: gates in LDS -> elementwise LSTM cell update.
template <int F, int Cin>
__global__ void __launch_bounds__(256)
lstm_step_wmma(const float* __restrict__ xin,      // (B,T,H,W,Cin)
               const float* __restrict__ hprev,    // (B,H,W,F)
               const _Float16* __restrict__ pw,    // prepacked (KP/32,NT,32,16)
               const float* __restrict__ bias,     // (G)
               float* __restrict__ cst,            // (B,H,W,F) in/out
               float* __restrict__ hnext,          // (B,H,W,F)
               float* __restrict__ hseq,           // (B,T,H,W,2F)
               int H, int W, int t, int dirOff) {
    constexpr int G   = 4 * F;
    constexpr int NT  = G / 16;
    constexpr int K1  = 9 * Cin;
    constexpr int K1P = (K1 + 31) & ~31;
    constexpr int K2  = 9 * F;
    constexpr int K2P = (K2 + 31) & ~31;
    constexpr int KP  = K1P + K2P;
    constexpr int XV  = (Cin % 4 == 0) ? 4 : ((Cin % 2 == 0) ? 2 : 1);
    constexpr int HV  = (F % 4 == 0) ? 4 : 2;

    __shared__ alignas(16) _Float16 lds_a[16 * KP];
    __shared__ float zbuf[16 * G];

    const int HW = H * W;
    const int m0 = blockIdx.x * 16;

    // ---------------- phase 1: im2col staging (16 rows x KP cols, f16)
    for (int idx = threadIdx.x; idx < 16 * 9; idx += blockDim.x) {
        int m = idx / 9, tap = idx - m * 9;
        int p = m0 + m;
        int b_ = p / HW; int r = p - b_ * HW; int y = r / W; int xw = r - y * W;
        int yy = y + tap / 3 - 1, xx = xw + tap % 3 - 1;
        bool inb = (yy >= 0) & (yy < H) & (xx >= 0) & (xx < W);
        // x-patch
        stage_row<Cin, XV>(&lds_a[m * KP + tap * Cin],
                           xin + (((long)b_ * T_SZ + t) * HW + (long)yy * W + xx) * Cin,
                           inb);
        // h-patch
        stage_row<F, HV>(&lds_a[m * KP + K1P + tap * F],
                         hprev + ((long)b_ * HW + (long)yy * W + xx) * F,
                         inb);
    }
    // zero-fill K padding columns
    if (K1P > K1) {
        constexpr int PADX = K1P - K1;
        for (int idx = threadIdx.x; idx < 16 * PADX; idx += blockDim.x) {
            int m = idx / PADX, k = idx - m * PADX;
            lds_a[m * KP + K1 + k] = (_Float16)0.f;
        }
    }
    if (K2P > K2) {
        constexpr int PADH = K2P - K2;
        for (int idx = threadIdx.x; idx < 16 * PADH; idx += blockDim.x) {
            int m = idx / PADH, k = idx - m * PADH;
            lds_a[m * KP + K1P + K2 + k] = (_Float16)0.f;
        }
    }
    __syncthreads();

    // ---------------- phase 2: GEMM, one 16x16 tile per wave
    const int wave  = threadIdx.x >> 5;
    const int lane  = threadIdx.x & 31;
    const int khalf = lane >> 4;
    const int ncol  = lane & 15;
    const int arow  = lane & 15;

    v8f acc = {};
    const _Float16* arowp = &lds_a[arow * KP];
    const _Float16* bbase = pw + ((size_t)wave * 32 + lane) * 16;
#pragma unroll 4
    for (int kc = 0; kc < KP / 32; kc++) {
        const int k0 = kc * 32;
        v8h alo = *reinterpret_cast<const v8h*>(arowp + k0 + 16 * khalf);
        v8h ahi = *reinterpret_cast<const v8h*>(arowp + k0 + 16 * khalf + 8);
        const _Float16* bp = bbase + (size_t)kc * NT * 32 * 16;
        v8h blo = *reinterpret_cast<const v8h*>(bp);
        v8h bhi = *reinterpret_cast<const v8h*>(bp + 8);
        v16h a, b;
#pragma unroll
        for (int j = 0; j < 8; j++) {
            a[j] = alo[j]; a[j + 8] = ahi[j];
            b[j] = blo[j]; b[j + 8] = bhi[j];
        }
        acc = __builtin_amdgcn_wmma_f32_16x16x32_f16(false, a, false, b,
                                                     (short)0, acc, false, false);
    }

    // stage z (+bias) in LDS; C/D layout: VGPR v -> row v+8*khalf, col ncol
    float bv = bias[wave * 16 + ncol];
#pragma unroll
    for (int v = 0; v < 8; v++)
        zbuf[(v + 8 * khalf) * G + wave * 16 + ncol] = acc[v] + bv;
    __syncthreads();

    // ---------------- phase 3: elementwise LSTM update
    for (int e = threadIdx.x; e < 16 * F; e += blockDim.x) {
        int m = e / F, ch = e - m * F;
        int pp = m0 + m;
        float zi = zbuf[m * G + ch];
        float zf = zbuf[m * G + F + ch];
        float zg = zbuf[m * G + 2 * F + ch];
        float zo = zbuf[m * G + 3 * F + ch];
        float ig = fmaxf(zi, 0.f);
        float fg = fmaxf(zf, 0.f);
        float gg = tanhf(zg);
        float og = fmaxf(zo, 0.f);
        long ci = (long)pp * F + ch;
        float cc = fg * cst[ci] + ig * gg;
        cst[ci] = cc;
        float hh = og * tanhf(cc);
        hnext[ci] = hh;
        int b_ = pp / HW; int r = pp - b_ * HW; int y = r / W; int x = r - y * W;
        hseq[((((long)b_ * T_SZ + t) * H + y) * (long)W + x) * (2 * F) + dirOff + ch] = hh;
    }
}

// ---------------------------------------------------------------- host side
template <int F, int Cin>
static void run_scan(const float* xin, int H, int W,
                     const _Float16* pw, const float* bias,
                     float* hA, float* hB, float* cbuf, float* hseq,
                     int dirOff, bool backward, hipStream_t stream,
                     float** hFinal) {
    const int totalM = B_SZ * H * W;                 // multiple of 16 for all layers
    const long stateN = (long)totalM * F;
    int zg = (int)((stateN + 255) / 256);
    zero_kernel<<<zg, 256, 0, stream>>>(hA, stateN);
    zero_kernel<<<zg, 256, 0, stream>>>(cbuf, stateN);
    float* bufs[2] = { hA, hB };
    int cur = 0;
    const int blocks = totalM / 16;
    const int thr = 32 * (4 * F / 16);
    for (int s = 0; s < T_SZ; s++) {
        int t = backward ? (T_SZ - 1 - s) : s;
        lstm_step_wmma<F, Cin><<<blocks, thr, 0, stream>>>(
            xin, bufs[cur], pw, bias, cbuf, bufs[1 - cur], hseq, H, W, t, dirOff);
        cur ^= 1;
    }
    *hFinal = bufs[cur];
}

template <int F, int Cin>
static _Float16* make_packed(const float* wx, const float* wh,
                             char* ws, size_t& off, hipStream_t stream) {
    constexpr int G   = 4 * F;
    constexpr int K1P = (9 * Cin + 31) & ~31;
    constexpr int K2P = (9 * F + 31) & ~31;
    constexpr int KP  = K1P + K2P;
    size_t elems = (size_t)KP * G;
    _Float16* pw = (_Float16*)(ws + off);
    off = (off + elems * sizeof(_Float16) + 255) & ~(size_t)255;
    int threads = (KP / 32) * (G / 16) * 32;
    prepack_weights<F, Cin><<<(threads + 255) / 256, 256, 0, stream>>>(wx, wh, pw);
    return pw;
}

extern "C" void kernel_launch(void* const* d_in, const int* in_sizes, int n_in,
                              void* d_out, int out_size, void* d_ws, size_t ws_size,
                              hipStream_t stream) {
    const float* x = (const float*)d_in[0];
    const float* w32[12]; const float* bvec[6];
    for (int i = 0; i < 6; i++) {
        w32[2 * i]     = (const float*)d_in[1 + 3 * i];   // Wx
        w32[2 * i + 1] = (const float*)d_in[2 + 3 * i];   // Wh
        bvec[i]        = (const float*)d_in[3 + 3 * i];   // b
    }
    const float *bn1g = (const float*)d_in[19], *bn1b = (const float*)d_in[20];
    const float *bn1m = (const float*)d_in[21], *bn1v = (const float*)d_in[22];
    const float *bn2g = (const float*)d_in[23], *bn2b = (const float*)d_in[24];
    const float *bn2m = (const float*)d_in[25], *bn2v = (const float*)d_in[26];
    float* out = (float*)d_out;

    char* ws = (char*)d_ws; size_t off = 0;
    auto alloc = [&](size_t bytes) -> void* {
        void* pp = ws + off; off = (off + bytes + 255) & ~(size_t)255; return pp;
    };

    // prepacked fragment-order weights (Wx ++ Wh, zero-padded chunks)
    _Float16* pw1f = make_packed<32, 6 >(w32[0],  w32[1],  ws, off, stream);
    _Float16* pw1b = make_packed<32, 6 >(w32[2],  w32[3],  ws, off, stream);
    _Float16* pw2f = make_packed<32, 64>(w32[4],  w32[5],  ws, off, stream);
    _Float16* pw2b = make_packed<32, 64>(w32[6],  w32[7],  ws, off, stream);
    _Float16* pw3f = make_packed<12, 64>(w32[8],  w32[9],  ws, off, stream);
    _Float16* pw3b = make_packed<12, 64>(w32[10], w32[11], ws, off, stream);

    // activation buffers
    const long hseq1N = (long)B_SZ * T_SZ * 48 * 56 * 64;
    const long pool1N = (long)B_SZ * T_SZ * 24 * 28 * 64;
    const long hseq2N = (long)B_SZ * T_SZ * 24 * 28 * 64;
    const long pool2N = (long)B_SZ * T_SZ * 12 * 14 * 64;
    float* hseq1 = (float*)alloc(sizeof(float) * hseq1N);
    float* pool1 = (float*)alloc(sizeof(float) * pool1N);
    float* hseq2 = (float*)alloc(sizeof(float) * hseq2N);
    float* pool2 = (float*)alloc(sizeof(float) * pool2N);
    const long stateMax = (long)B_SZ * 48 * 56 * 32;
    float* hA = (float*)alloc(sizeof(float) * stateMax);
    float* hB = (float*)alloc(sizeof(float) * stateMax);
    float* cb = (float*)alloc(sizeof(float) * stateMax);
    float* hFin = nullptr;

    // ---------------- layer 1 : (8,24,48,56,6) -> hseq1 (…,64) -> pool1
    run_scan<32, 6>(x, 48, 56, pw1f, bvec[0], hA, hB, cb, hseq1, 0,  false, stream, &hFin);
    run_scan<32, 6>(x, 48, 56, pw1b, bvec[1], hA, hB, cb, hseq1, 32, true,  stream, &hFin);
    bnpool_kernel<<<(int)((pool1N + 255) / 256), 256, 0, stream>>>(
        hseq1, pool1, bn1g, bn1b, bn1m, bn1v, 48, 56, 64, 24, 28, pool1N);

    // ---------------- layer 2 : pool1 (8,24,24,28,64) -> hseq2 -> pool2
    run_scan<32, 64>(pool1, 24, 28, pw2f, bvec[2], hA, hB, cb, hseq2, 0,  false, stream, &hFin);
    run_scan<32, 64>(pool1, 24, 28, pw2b, bvec[3], hA, hB, cb, hseq2, 32, true,  stream, &hFin);
    bnpool_kernel<<<(int)((pool2N + 255) / 256), 256, 0, stream>>>(
        hseq2, pool2, bn2g, bn2b, bn2m, bn2v, 24, 28, 64, 12, 14, pool2N);

    // ---------------- layer 3 : pool2 (8,24,12,14,64) -> seq directly in d_out
    const long seqN = (long)B_SZ * T_SZ * 12 * 14 * 24;
    const long stN  = (long)B_SZ * 12 * 14 * 12;
    float* seqOut = out;
    float* fhOut = out + seqN;
    float* fcOut = fhOut + stN;
    float* bhOut = fcOut + stN;
    float* bcOut = bhOut + stN;

    run_scan<12, 64>(pool2, 12, 14, pw3f, bvec[4], hA, hB, cb, seqOut, 0, false, stream, &hFin);
    copy_kernel<<<(int)((stN + 255) / 256), 256, 0, stream>>>(fhOut, hFin, stN);
    copy_kernel<<<(int)((stN + 255) / 256), 256, 0, stream>>>(fcOut, cb, stN);

    run_scan<12, 64>(pool2, 12, 14, pw3b, bvec[5], hA, hB, cb, seqOut, 12, true, stream, &hFin);
    copy_kernel<<<(int)((stN + 255) / 256), 256, 0, stream>>>(bhOut, hFin, stN);
    copy_kernel<<<(int)((stN + 255) / 256), 256, 0, stream>>>(bcOut, cb, stN);

    (void)in_sizes; (void)n_in; (void)out_size; (void)ws_size;
}
